// CrissCrossAttention_35502199669040
// MI455X (gfx1250) — matrix operand order repaired and checked by reading, not compiled, not run on hardware
//
#include <hip/hip_runtime.h>
#include <hip/hip_bf16.h>
#include <stdint.h>

#define Bsz   4
#define Cch   512
#define CQch  64
#define Hdim  128
#define Wdim  128
#define NPIX  (Hdim * Wdim)   // 16384 pixels per batch

typedef __bf16 bf16_t;
typedef __attribute__((ext_vector_type(16))) __bf16        v16bf;
typedef __attribute__((ext_vector_type(8)))  __bf16        v8bf;
typedef __attribute__((ext_vector_type(8)))  float         v8f;
typedef __attribute__((ext_vector_type(4)))  unsigned int  v4u;
typedef __attribute__((ext_vector_type(8)))  int           v8i;
typedef __attribute__((ext_vector_type(4)))  int           v4i;

// ---------- bf16 conversion (truncation; accumulate is fp32) ----------
__device__ __forceinline__ bf16_t f2bf(float f) {
  union { float f; unsigned u; } c; c.f = f;
  union { unsigned short s; bf16_t b; } r;
  r.s = (unsigned short)(c.u >> 16);
  return r.b;
}
__device__ __forceinline__ v16bf cat8(v8bf lo, v8bf hi) {
  return __builtin_shufflevector(lo, hi, 0, 1, 2, 3, 4, 5, 6, 7, 8, 9, 10, 11,
                                 12, 13, 14, 15);
}
__device__ __forceinline__ v16bf pack16(float4 x0, float4 x1, float4 x2,
                                        float4 x3) {
  v16bf a;
  a[0] = f2bf(x0.x);  a[1] = f2bf(x0.y);  a[2] = f2bf(x0.z);  a[3] = f2bf(x0.w);
  a[4] = f2bf(x1.x);  a[5] = f2bf(x1.y);  a[6] = f2bf(x1.z);  a[7] = f2bf(x1.w);
  a[8] = f2bf(x2.x);  a[9] = f2bf(x2.y);  a[10] = f2bf(x2.z); a[11] = f2bf(x2.w);
  a[12] = f2bf(x3.x); a[13] = f2bf(x3.y); a[14] = f2bf(x3.z); a[15] = f2bf(x3.w);
  return a;
}

// ---------- WMMA fragment loaders (wave32; contiguous k order, same in A and B) ----------
// A 16x32 bf16: lane row M=lane&15, K in {kb..kb+15}, kb=(lane>>4)*16.
// Vectorized: two 16-byte LDS loads per lane (offsets are 16B-aligned by design).
__device__ __forceinline__ v16bf ldsA_bf_v(const bf16_t* s, int row0, int k0,
                                           int ld, int lane) {
  int m = lane & 15, kb = (lane >> 4) * 16;
  const bf16_t* p = s + (size_t)(row0 + m) * ld + k0 + kb;
  return cat8(*(const v8bf*)p, *(const v8bf*)(p + 8));
}
// B 32x16 bf16 from TRANSPOSED LDS storage [N][K] (K contiguous per lane):
// lane col N=lane&15, kb=(lane>>4)*16 -> one 32-byte contiguous span.
__device__ __forceinline__ v16bf ldsBt_bf_v(const bf16_t* s, int col0, int k0,
                                            int ld, int lane) {
  int n = lane & 15, kb = (lane >> 4) * 16;
  const bf16_t* p = s + (size_t)(col0 + n) * ld + k0 + kb;
  return cat8(*(const v8bf*)p, *(const v8bf*)(p + 8));
}
// B 32x16 from transposed fp32 LDS storage [N][K]: 4x float4 + convert.
__device__ __forceinline__ v16bf ldsBt_f32_v(const float* s, int col0, int k0,
                                             int ld, int lane) {
  int n = lane & 15, kb = (lane >> 4) * 16;
  const float* p = s + (size_t)(col0 + n) * ld + k0 + kb;
  return pack16(*(const float4*)p, *(const float4*)(p + 4),
                *(const float4*)(p + 8), *(const float4*)(p + 12));
}
// B 32x16 from row-major fp32 LDS [K][N] (strided; used for the TDM X tile).
__device__ __forceinline__ v16bf ldsB_f32(const float* s, int k0, int col0,
                                          int ld, int lane) {
  int n = lane & 15, kb = (lane >> 4) * 16;
  const float* p = s + (size_t)(k0 + kb) * ld + col0 + n;
  v16bf b;
#pragma unroll
  for (int i = 0; i < 16; ++i) b[i] = f2bf(p[(size_t)i * ld]);
  return b;
}
// A 16x32 direct from GLOBAL fp32 row-major (weights): 4x global_load_b128.
__device__ __forceinline__ v16bf gA_f32(const float* base, int ldg, int lane) {
  int m = lane & 15, kb = (lane >> 4) * 16;
  const float* p = base + (size_t)m * ldg + kb;
  return pack16(*(const float4*)p, *(const float4*)(p + 4),
                *(const float4*)(p + 8), *(const float4*)(p + 12));
}

__device__ __forceinline__ v8f wmma_bf16(v16bf a, v16bf b, v8f c) {
  return __builtin_amdgcn_wmma_f32_16x16x32_bf16(
      /*neg_a=*/false, a, /*neg_b=*/false, b,
      /*c_mod=*/(short)0, c, /*reuse_a=*/false, /*reuse_b=*/false);
}

// ---------- Tensor Data Mover: 2D tile (tile1 rows x tile0 elems, 4B) ----------
// D# per CDNA5 ISA. 6-arg builtin on this toolchain:
//   (uint32x4 g0, int32x8 g1, int32x4 g2, int32x4 g3, int32x8 g4, i32 cpol)
__device__ __forceinline__ void tdm_load_2d(const void* gptr, unsigned ldsOff,
                                            unsigned td0, unsigned td1,
                                            unsigned tile0, unsigned tile1,
                                            unsigned long long stride0) {
  unsigned long long ga = (unsigned long long)(uintptr_t)gptr;
  v4u g0;
  g0[0] = 1u;                                           // count=1, user desc
  g0[1] = ldsOff;                                       // lds_addr (bytes)
  g0[2] = (unsigned)(ga & 0xFFFFFFFFull);
  g0[3] = (unsigned)((ga >> 32) & 0x01FFFFFFull) | (2u << 30); // type=2
  v8i g1;
  g1[0] = (int)(2u << 16);                              // data_size=2 -> 4B
  g1[1] = (int)((td0 & 0xFFFFu) << 16);                 // tensor_dim0 lo16
  g1[2] = (int)(((td0 >> 16) & 0xFFFFu) | ((td1 & 0xFFFFu) << 16));
  g1[3] = (int)(((td1 >> 16) & 0xFFFFu) | ((tile0 & 0xFFFFu) << 16));
  g1[4] = (int)(tile1 & 0xFFFFu);                       // tile_dim1, tile_dim2=0
  g1[5] = (int)(unsigned)(stride0 & 0xFFFFFFFFull);     // dim0 stride lo32
  g1[6] = (int)(unsigned)((stride0 >> 32) & 0xFFFFull); // dim0 stride hi16
  g1[7] = 0;
  v4i z4 = {0, 0, 0, 0};
  v8i z8 = {0, 0, 0, 0, 0, 0, 0, 0};
  __builtin_amdgcn_tensor_load_to_lds(g0, g1, z4, z4, z8, 0);
}

// =====================================================================
// Kernel 1: 1x1-conv projection as GEMM  out(Cout x NPIX) = W x X + b
// grid = (NPIX/128, Cout/16, B), block = 256 (8 wave32s).
// X tiles double-buffered via TDM; W fragments read straight from L2.
// =====================================================================
__global__ __launch_bounds__(256) void proj_kernel(
    const float* __restrict__ X, const float* __restrict__ Wm,
    const float* __restrict__ bias, float* __restrict__ out, int Cout) {
  __shared__ __align__(16) float lds_x[2][32 * 128];  // double-buffered TDM dest

  const int tid = threadIdx.x;
  const int lane = tid & 31, wave = tid >> 5;
  const int pix0 = blockIdx.x * 128;
  const int oc0  = blockIdx.y * 16;
  const int b    = blockIdx.z;
  const float* Xb = X + (size_t)b * Cch * NPIX;
  const float* Wb = Wm + (size_t)oc0 * Cch;

  v8f acc = {0.f, 0.f, 0.f, 0.f, 0.f, 0.f, 0.f, 0.f};

  if (wave == 0) {
    tdm_load_2d(Xb + pix0, (unsigned)(uintptr_t)(void*)lds_x[0],
                NPIX, Cch, 128, 32, NPIX);
  }
  for (int it = 0; it < Cch / 32; ++it) {
    if (wave == 0) {
      if (it + 1 < Cch / 32) {
        tdm_load_2d(Xb + (size_t)(it + 1) * 32 * NPIX + pix0,
                    (unsigned)(uintptr_t)(void*)lds_x[(it + 1) & 1],
                    NPIX, Cch, 128, 32, NPIX);
        __builtin_amdgcn_s_wait_tensorcnt(1);  // tile `it` has landed
      } else {
        __builtin_amdgcn_s_wait_tensorcnt(0);
      }
    }
    __syncthreads();

    v16bf a  = gA_f32(Wb + it * 32, Cch, lane);
    v16bf bb = ldsB_f32(lds_x[it & 1], 0, wave * 16, 128, lane);
    acc = wmma_bf16(a, bb, acc);
    __syncthreads();  // all waves done with buffer before TDM reuses it
  }

  int n = lane & 15;
  int mhi = (lane >> 4) * 8;
  int pix = pix0 + wave * 16 + n;
#pragma unroll
  for (int r = 0; r < 8; ++r) {
    int m = mhi + r;  // D layout: VGPR r holds M=r (lanes 0-15) / M=r+8
    out[((size_t)b * Cout + oc0 + m) * NPIX + pix] = acc[r] + bias[oc0 + m];
  }
}

// =====================================================================
// Kernel 2: one attention line (row or column). grid = 512, block = 256.
// softmax(Q Kt) V for one (b, h) or (b, w) line; L=128, CQ=64.
// =====================================================================
__global__ __launch_bounds__(256) void attn_kernel(
    const float* __restrict__ q, const float* __restrict__ k,
    const float* __restrict__ v, float* __restrict__ outc, int vertical) {
  __shared__ __align__(16) char smem[32768 + 2 * 16 * 128 * 4];  // 48 KB
  bf16_t* qs  = (bf16_t*)smem;             // [128][64]  Q  (row, a)
  bf16_t* ks  = (bf16_t*)(smem + 16384);   // [128][64]  Kt transposed (j, a)
  bf16_t* as_ = (bf16_t*)smem;             // [128][128] A  (reuses Q/K)
  float*  vsf = (float*)(smem + 32768);    // [2][16][128] V tiles (fp32)

  const int tid = threadIdx.x, lane = tid & 31, wave = tid >> 5;
  const int blk = blockIdx.x;
  const int b  = blk >> 7;
  const int rc = blk & 127;  // h (horizontal) or w (vertical)

  const float* qb = q + (size_t)b * CQch * NPIX;
  const float* kb = k + (size_t)b * CQch * NPIX;
  const float* vb = v + (size_t)b * Cch * NPIX;

  // ---- stage Q: qs[row][a] (horizontal path uses float4 global loads) ----
  for (int e4 = tid; e4 < 128 * CQch / 4; e4 += 256) {
    int a = e4 >> 5;             // 32 row-quads per channel
    int row = (e4 & 31) * 4;
    if (!vertical) {
      float4 vq = *(const float4*)(qb + (size_t)(a * Hdim + rc) * Wdim + row);
      qs[(row + 0) * CQch + a] = f2bf(vq.x);
      qs[(row + 1) * CQch + a] = f2bf(vq.y);
      qs[(row + 2) * CQch + a] = f2bf(vq.z);
      qs[(row + 3) * CQch + a] = f2bf(vq.w);
    } else {  // faithful raw reshape of (Cq,H) -> (H,Cq)
#pragma unroll
      for (int i = 0; i < 4; ++i) {
        int f = (row + i) * CQch + a;
        qs[(row + i) * CQch + a] =
            f2bf(qb[(size_t)((f >> 7) * Hdim + (f & 127)) * Wdim + rc]);
      }
    }
  }
  // ---- stage Kt TRANSPOSED: ks[j][a] ----
  for (int e = tid; e < CQch * 128; e += 256) {
    int a = e >> 7, j = e & 127;
    float val;
    if (!vertical) {  // faithful raw reshape of (W,Cq) -> (Cq,W)
      val = kb[(size_t)((j & 63) * Hdim + rc) * Wdim + (2 * a + (j >> 6))];
    } else {
      val = kb[(size_t)(a * Hdim + j) * Wdim + rc];
    }
    ks[j * CQch + a] = f2bf(val);
  }
  __syncthreads();

  // ---- logits: 16x128 per wave, K = 64 ----
  const int m0 = wave * 16;
  v16bf qa0 = ldsA_bf_v(qs, m0, 0, CQch, lane);
  v16bf qa1 = ldsA_bf_v(qs, m0, 32, CQch, lane);
  v8f acc[8];
#pragma unroll
  for (int nt = 0; nt < 8; ++nt) {
    v8f z = {0.f, 0.f, 0.f, 0.f, 0.f, 0.f, 0.f, 0.f};
    z = wmma_bf16(qa0, ldsBt_bf_v(ks, nt * 16, 0, CQch, lane), z);
    acc[nt] = wmma_bf16(qa1, ldsBt_bf_v(ks, nt * 16, 32, CQch, lane), z);
  }

  // ---- register softmax per row (16-lane half-group owns a row) ----
#pragma unroll
  for (int r = 0; r < 8; ++r) {
    float mx = -3.0e38f;
#pragma unroll
    for (int nt = 0; nt < 8; ++nt) mx = fmaxf(mx, acc[nt][r]);
    for (int sh = 1; sh <= 8; sh <<= 1) mx = fmaxf(mx, __shfl_xor(mx, sh, 32));
    float sm = 0.f;
#pragma unroll
    for (int nt = 0; nt < 8; ++nt) {
      float e = __expf(acc[nt][r] - mx);
      acc[nt][r] = e;
      sm += e;
    }
    for (int sh = 1; sh <= 8; sh <<= 1) sm += __shfl_xor(sm, sh, 32);
    float inv = 1.0f / sm;
#pragma unroll
    for (int nt = 0; nt < 8; ++nt) acc[nt][r] *= inv;
  }
  __syncthreads();  // all waves done reading qs/ks before aliasing as as_

  {
    int n = lane & 15, mhi = (lane >> 4) * 8;
#pragma unroll
    for (int nt = 0; nt < 8; ++nt)
#pragma unroll
      for (int r = 0; r < 8; ++r)
        as_[(size_t)(m0 + mhi + r) * 128 + nt * 16 + n] = f2bf(acc[nt][r]);
  }
  __syncthreads();

  // A-fragments reused across all 32 channel tiles
  v16bf af[4];
#pragma unroll
  for (int t = 0; t < 4; ++t) af[t] = ldsA_bf_v(as_, m0, t * 32, 128, lane);

  // ---- out = A (128x128) x V (128x512), 16 channels per iteration ----
  // Horizontal: V tiles are contiguous rows -> pipelined TDM double-buffer.
  if (!vertical && wave == 0) {
    tdm_load_2d(vb + (size_t)rc * Wdim, (unsigned)(uintptr_t)(void*)vsf,
                Wdim, Cch * Hdim, 128, 16, NPIX);
  }
  for (int it = 0; it < Cch / 16; ++it) {
    const int c0 = it * 16;
    float* cur = vsf + (it & 1) * (16 * 128);
    if (!vertical) {
      if (wave == 0) {
        if (it + 1 < Cch / 16) {
          tdm_load_2d(vb + ((size_t)(c0 + 16) * Hdim + rc) * Wdim,
                      (unsigned)(uintptr_t)(void*)(vsf + ((it + 1) & 1) * (16 * 128)),
                      Wdim, Cch * Hdim, 128, 16, NPIX);
          __builtin_amdgcn_s_wait_tensorcnt(1);
        } else {
          __builtin_amdgcn_s_wait_tensorcnt(0);
        }
      }
    } else {  // faithful raw reshape of (C,H) -> (H,C): scalar gather
      for (int e = tid; e < 16 * 128; e += 256) {
        int ci = e >> 7, j = e & 127;
        int f = j * Cch + c0 + ci;
        cur[ci * 128 + j] =
            vb[(size_t)((f >> 7) * Hdim + (f & 127)) * Wdim + rc];
      }
    }
    __syncthreads();

    v8f accv = {0.f, 0.f, 0.f, 0.f, 0.f, 0.f, 0.f, 0.f};
#pragma unroll
    for (int t = 0; t < 4; ++t)
      accv = wmma_bf16(af[t], ldsBt_f32_v(cur, 0, t * 32, 128, lane), accv);

    int n = lane & 15, mhi = (lane >> 4) * 8;
#pragma unroll
    for (int r = 0; r < 8; ++r) {
      int m = m0 + mhi + r;
      size_t addr;
      if (!vertical)
        addr = (((size_t)b * Cch + c0 + n) * Hdim + rc) * Wdim + m;
      else
        addr = (((size_t)b * Cch + c0 + n) * Hdim + m) * Wdim + rc;
      outc[addr] = accv[r];
    }
    __syncthreads();  // before the buffer is overwritten
  }
}

// =====================================================================
// Kernel 3: out = gamma * (ch + cv) + x   (float4 vectorized)
// =====================================================================
__global__ __launch_bounds__(256) void combine_kernel(
    const float4* __restrict__ ch, const float4* __restrict__ cv,
    const float4* __restrict__ x, const float* __restrict__ gamma,
    float4* __restrict__ out) {
  size_t idx = (size_t)blockIdx.x * 256 + threadIdx.x;
  __builtin_prefetch(&x[idx + 4096], 0, 1);  // global_prefetch_b8
  float g = gamma[0];
  float4 a = ch[idx], c = cv[idx], xx = x[idx];
  float4 o;
  o.x = g * (a.x + c.x) + xx.x;
  o.y = g * (a.y + c.y) + xx.y;
  o.z = g * (a.z + c.z) + xx.z;
  o.w = g * (a.w + c.w) + xx.w;
  out[idx] = o;
}

// =====================================================================
extern "C" void kernel_launch(void* const* d_in, const int* in_sizes, int n_in,
                              void* d_out, int out_size, void* d_ws,
                              size_t ws_size, hipStream_t stream) {
  const float* x     = (const float*)d_in[0];
  const float* Wq    = (const float*)d_in[1];
  const float* bq    = (const float*)d_in[2];
  const float* Wk    = (const float*)d_in[3];
  const float* bk    = (const float*)d_in[4];
  const float* Wv    = (const float*)d_in[5];
  const float* bv    = (const float*)d_in[6];
  const float* gamma = (const float*)d_in[7];
  float* out = (float*)d_out;

  // Workspace layout (floats): q | k | v | ch | cv
  float* ws = (float*)d_ws;
  float* q  = ws;                              // 4*64*16384  =  4,194,304
  float* kk = ws + (size_t)4194304;            // 4,194,304
  float* v  = ws + (size_t)8388608;            // 4*512*16384 = 33,554,432
  float* ch = ws + (size_t)41943040;           // 33,554,432
  float* cv = ws + (size_t)75497472;           // 33,554,432

  dim3 blk(256);
  // Projections: out channels 64 / 64 / 512
  proj_kernel<<<dim3(NPIX / 128, CQch / 16, Bsz), blk, 0, stream>>>(x, Wq, bq, q, CQch);
  proj_kernel<<<dim3(NPIX / 128, CQch / 16, Bsz), blk, 0, stream>>>(x, Wk, bk, kk, CQch);
  proj_kernel<<<dim3(NPIX / 128, Cch / 16, Bsz), blk, 0, stream>>>(x, Wv, bv, v, Cch);
  // Row then column attention
  attn_kernel<<<dim3(Bsz * Hdim), blk, 0, stream>>>(q, kk, v, ch, 0);
  attn_kernel<<<dim3(Bsz * Wdim), blk, 0, stream>>>(q, kk, v, cv, 1);
  // Final combine
  combine_kernel<<<dim3((Bsz * Cch * NPIX) / 4 / 256), blk, 0, stream>>>(
      (const float4*)ch, (const float4*)cv, (const float4*)x, gamma,
      (float4*)out);
}
